// WaveRNN_55327768708091
// MI455X (gfx1250) — matrix operand order repaired and checked by reading, not compiled, not run on hardware
//
#include <hip/hip_runtime.h>
#include <hip/hip_bf16.h>
#include <math.h>

// Problem dims (WaveRNN-style GRU cell + dual 2-layer MLP heads)
#define B_DIM 32768
#define H_DIM 896
#define S_DIM 448
#define Q_DIM 256
#define KB_GRU (H_DIM / 32)   // 28 K-tiles of 32
#define KB2_GRU (H_DIM / 64)  // 14 K-tiles of 64

typedef __attribute__((ext_vector_type(16))) __bf16 v16bf;
typedef __attribute__((ext_vector_type(8)))  float  v8f;
typedef __attribute__((ext_vector_type(4)))  int    v4i_t;

typedef __attribute__((address_space(1))) v4i_t g_v4i;  // global int4
typedef __attribute__((address_space(3))) v4i_t l_v4i;  // LDS int4

// ---------------------------------------------------------------------------
// Async global->LDS b128 (CDNA5 ASYNCcnt path): (int4 AS1*, int4 AS3*, imm, imm)
// ---------------------------------------------------------------------------
__device__ __forceinline__ void async_copy_b128(const __bf16* g, __bf16* l) {
#if __has_builtin(__builtin_amdgcn_global_load_async_to_lds_b128)
  g_v4i* gp = (g_v4i*)(unsigned long long)g;
  l_v4i* lp = (l_v4i*)(unsigned int)(unsigned long long)l;
  __builtin_amdgcn_global_load_async_to_lds_b128(gp, lp, 0, 0);
#else
  *(uint4*)l = *(const uint4*)g;
#endif
}
__device__ __forceinline__ void async_wait0() {
#if __has_builtin(__builtin_amdgcn_s_wait_asynccnt)
  __builtin_amdgcn_s_wait_asynccnt(0);
#endif
}

// ---------------------------------------------------------------------------
// Fragment loads, forced to b128 LDS reads.
// A: 16x32 bf16, lane L: M=L%16, two 8-element K runs selected by L/16.
// B: pre-swizzled so lane L's 16 K-ascending elements are contiguous (32B).
// ---------------------------------------------------------------------------
struct V32B { uint4 lo, hi; };

__device__ __forceinline__ v16bf frag_a(const __bf16* As, int m_base, int lane) {
  const int m  = m_base + (lane & 15);
  const int kh = (lane >> 4) << 4;
  const __bf16* p = As + m * 32 + kh;
  V32B v;
  v.lo = *(const uint4*)p;         // K = kh .. kh+7
  v.hi = *(const uint4*)(p + 8);   // K = kh+8 .. kh+15
  return __builtin_bit_cast(v16bf, v);
}

__device__ __forceinline__ v16bf frag_b_sw(const __bf16* frag_base, int lane) {
  const __bf16* p = frag_base + lane * 16;
  V32B v;
  v.lo = *(const uint4*)p;
  v.hi = *(const uint4*)(p + 8);
  return __builtin_bit_cast(v16bf, v);
}

__device__ __forceinline__ float sigmoidf_(float x) { return 1.0f / (1.0f + __expf(-x)); }

__device__ __forceinline__ float tanh_fast(float x) {
#if __has_builtin(__builtin_amdgcn_tanhf)
  return __builtin_amdgcn_tanhf(x);   // gfx1250 V_TANH_F32
#else
  return tanhf(x);
#endif
}

// ---------------------------------------------------------------------------
// f32 -> bf16, 4 elements/thread
// ---------------------------------------------------------------------------
__global__ void cvt_f32_bf16(const float* __restrict__ in, __bf16* __restrict__ out, int n4) {
  const int i = blockIdx.x * blockDim.x + threadIdx.x;
  if (i < n4) {
    const float4 v = ((const float4*)in)[i];
    __bf16 o[4] = {(__bf16)v.x, (__bf16)v.y, (__bf16)v.z, (__bf16)v.w};
    ((uint2*)out)[i] = *(uint2*)o;
  }
}

// ---------------------------------------------------------------------------
// Weight swizzle: W[K,N] f32 row-major -> bf16 fragment-native layout
//   out[((nb*KB + kb)*32 + lane)*16 + j] = W[kb*32 + (lane/16)*16 + j][nb*16 + lane%16]
// ---------------------------------------------------------------------------
__global__ void swizzle_w_bf16(const float* __restrict__ in, __bf16* __restrict__ out,
                               int K, int N) {
  const int t = blockIdx.x * blockDim.x + threadIdx.x;
  const int KB = K >> 5;
  const int NB = N >> 4;
  if (t >= NB * KB * 32) return;
  const int lane = t & 31;
  const int kb = (t >> 5) % KB;
  const int nb = (t >> 5) / KB;
  const int n  = nb * 16 + (lane & 15);
  const int k0 = kb * 32 + ((lane >> 4) << 4);
  __bf16* o = out + (size_t)t * 16;
#pragma unroll
  for (int j = 0; j < 16; ++j) o[j] = (__bf16)in[(size_t)(k0 + j) * N + n];
}

// ---------------------------------------------------------------------------
// Fused GRU GEMM. Block tile [64 rows x 64 hidden cols]; u/r/e column tiles
// accumulated together -> GRU pointwise epilogue in-register. 8 waves:
// (wm 0..1, wn 0..3), 2 M-subtiles x 3 gates = 6 accumulators each.
// Double-buffered LDS, BK=64 per stage (12 back-to-back WMMAs per wave per
// stage, one wait+barrier per 64 K).
// ---------------------------------------------------------------------------
__global__ __launch_bounds__(256) void gru_gemm(
    const __bf16* __restrict__ A,    // [B,H] prev_hidden bf16
    const __bf16* __restrict__ Wsw,  // swizzled R_W, NB = 3H/16, KB = 28
    const float*  __restrict__ prev_y,
    const float*  __restrict__ cur_c,
    const float*  __restrict__ IcW,
    const float*  __restrict__ IfW,
    const float*  __restrict__ ph_f32,
    const float*  __restrict__ bu,
    const float*  __restrict__ br,
    const float*  __restrict__ be,
    float*        __restrict__ hidden_f32,
    __bf16*       __restrict__ hidden_b16)
{
  __shared__ __bf16 As[2][2][64 * 32];   // buf x k-half : 16KB
  __shared__ __bf16 Bs[2][2][12 * 512];  // buf x k-half x (3 gates x 4 wn) : 48KB

  const int tid  = threadIdx.x;
  const int lane = tid & 31;
  const int wave = tid >> 5;
  const int wm   = wave & 1;
  const int wn   = wave >> 1;
  const int row0 = blockIdx.x * 64;
  const int n0   = blockIdx.y * 64;

  v8f acc[2][3] = {};

  auto stage = [&](int buf, int kb2) {
    // A: 64 rows x 64 K = 512 x 16B, split into two 32-K halves
#pragma unroll
    for (int r = 0; r < 2; ++r) {
      const int i  = tid + 256 * r;
      const int ar = i >> 3;
      const int c8 = (i & 7) * 8;
      const int h  = c8 >> 5;
      const int cw = c8 & 31;
      async_copy_b128(A + (size_t)(row0 + ar) * H_DIM + kb2 * 64 + c8,
                      &As[buf][h][ar * 32 + cw]);
    }
    // B: 2 halves x 12 chunks x 512 elems = 1536 x 16B
#pragma unroll
    for (int r = 0; r < 6; ++r) {
      const int i     = tid + 256 * r;
      const int h     = i / 768;
      const int rem   = i % 768;
      const int chunk = rem >> 6;
      const int off   = (rem & 63) * 8;
      const int s     = chunk >> 2;
      const int wc    = chunk & 3;
      const int nb    = s * (H_DIM / 16) + n0 / 16 + wc;
      async_copy_b128(Wsw + ((size_t)nb * KB_GRU + kb2 * 2 + h) * 512 + off,
                      &Bs[buf][h][chunk * 512 + off]);
    }
  };

  stage(0, 0);
  async_wait0();
  __syncthreads();

  for (int kb2 = 0; kb2 < KB2_GRU; ++kb2) {
    const int cur = kb2 & 1;
    if (kb2 + 1 < KB2_GRU) stage(1 - cur, kb2 + 1);

#pragma unroll
    for (int h = 0; h < 2; ++h) {
      const v16bf a0 = frag_a(&As[cur][h][0], wm * 32 + 0, lane);
      const v16bf a1 = frag_a(&As[cur][h][0], wm * 32 + 16, lane);
#pragma unroll
      for (int s = 0; s < 3; ++s) {
        const v16bf b = frag_b_sw(&Bs[cur][h][(s * 4 + wn) * 512], lane);
        acc[0][s] = __builtin_amdgcn_wmma_f32_16x16x32_bf16(false, a0, false, b, (short)0, acc[0][s], false, false);
        acc[1][s] = __builtin_amdgcn_wmma_f32_16x16x32_bf16(false, a1, false, b, (short)0, acc[1][s], false, false);
      }
    }
    if (kb2 + 1 < KB2_GRU) async_wait0();
    __syncthreads();
  }

  // Epilogue. C/D layout: element i -> M = i + 8*(lane/16), N = lane%16.
  const int col = n0 + wn * 16 + (lane & 15);
  const float bias_u_v = bu[col];
  const float bias_r_v = br[col];
  const float bias_e_v = be[col];
#pragma unroll
  for (int sub = 0; sub < 2; ++sub) {
#pragma unroll
    for (int i = 0; i < 8; ++i) {
      const int row = row0 + wm * 32 + sub * 16 + ((lane >> 4) << 3) + i;
      const float y0 = prev_y[row * 2 + 0];
      const float y1 = prev_y[row * 2 + 1];
      float Iu, Ir, Ie;
      if (col < S_DIM) {
        const int c = col;
        Iu = y0 * IcW[0 * 3 * S_DIM + c] + y1 * IcW[1 * 3 * S_DIM + c];
        Ir = y0 * IcW[0 * 3 * S_DIM + S_DIM + c] + y1 * IcW[1 * 3 * S_DIM + S_DIM + c];
        Ie = y0 * IcW[0 * 3 * S_DIM + 2 * S_DIM + c] + y1 * IcW[1 * 3 * S_DIM + 2 * S_DIM + c];
      } else {
        const int c = col - S_DIM;
        const float cc = cur_c[row];
        Iu = y0 * IfW[0 * 3 * S_DIM + c] + y1 * IfW[1 * 3 * S_DIM + c] + cc * IfW[2 * 3 * S_DIM + c];
        Ir = y0 * IfW[0 * 3 * S_DIM + S_DIM + c] + y1 * IfW[1 * 3 * S_DIM + S_DIM + c] +
             cc * IfW[2 * 3 * S_DIM + S_DIM + c];
        Ie = y0 * IfW[0 * 3 * S_DIM + 2 * S_DIM + c] + y1 * IfW[1 * 3 * S_DIM + 2 * S_DIM + c] +
             cc * IfW[2 * 3 * S_DIM + 2 * S_DIM + c];
      }
      const float u = sigmoidf_(acc[sub][0][i] + Iu + bias_u_v);
      const float r = sigmoidf_(acc[sub][1][i] + Ir + bias_r_v);
      const float e = tanh_fast(r * acc[sub][2][i] + Ie + bias_e_v);
      const float h = u * ph_f32[(size_t)row * H_DIM + col] + (1.0f - u) * e;
      hidden_f32[(size_t)row * H_DIM + col] = h;
      hidden_b16[(size_t)row * H_DIM + col] = (__bf16)h;
    }
  }
}

// ---------------------------------------------------------------------------
// MLP GEMM: out = act(A @ W + bias). Block tile [64 x 64], BK=64 stages.
// relu_to_bf16=1: ReLU -> bf16 scratch ; 0: f32 final output.
// ---------------------------------------------------------------------------
__global__ __launch_bounds__(256) void mlp_gemm(
    const __bf16* __restrict__ A, int lda,
    const __bf16* __restrict__ Wsw, int KB2,  // K/64 ; KB = 2*KB2
    const float*  __restrict__ bias,
    int relu_to_bf16,
    __bf16* __restrict__ outb, int ldob,
    float*  __restrict__ outf, int ldof)
{
  __shared__ __bf16 As[2][2][64 * 32];  // 16KB
  __shared__ __bf16 Bs[2][2][4 * 512];  // 16KB : 4 wn chunks per half

  const int tid  = threadIdx.x;
  const int lane = tid & 31;
  const int wave = tid >> 5;
  const int wm   = wave & 1;
  const int wn   = wave >> 1;
  const int row0 = blockIdx.x * 64;
  const int n0   = blockIdx.y * 64;
  const int KB   = KB2 * 2;

  v8f acc[2] = {};

  auto stage = [&](int buf, int kb2) {
#pragma unroll
    for (int r = 0; r < 2; ++r) {      // A: 512 x 16B
      const int i  = tid + 256 * r;
      const int ar = i >> 3;
      const int c8 = (i & 7) * 8;
      const int h  = c8 >> 5;
      const int cw = c8 & 31;
      async_copy_b128(A + (size_t)(row0 + ar) * lda + kb2 * 64 + c8,
                      &As[buf][h][ar * 32 + cw]);
    }
#pragma unroll
    for (int r = 0; r < 2; ++r) {      // B: 2 halves x 4 chunks x 64 = 512 x 16B
      const int i     = tid + 256 * r;
      const int h     = i >> 8;
      const int rem   = i & 255;
      const int chunk = rem >> 6;
      const int off   = (rem & 63) * 8;
      const int nb    = n0 / 16 + chunk;
      async_copy_b128(Wsw + ((size_t)nb * KB + kb2 * 2 + h) * 512 + off,
                      &Bs[buf][h][chunk * 512 + off]);
    }
  };

  stage(0, 0);
  async_wait0();
  __syncthreads();

  for (int kb2 = 0; kb2 < KB2; ++kb2) {
    const int cur = kb2 & 1;
    if (kb2 + 1 < KB2) stage(1 - cur, kb2 + 1);

#pragma unroll
    for (int h = 0; h < 2; ++h) {
      const v16bf a0 = frag_a(&As[cur][h][0], wm * 32 + 0, lane);
      const v16bf a1 = frag_a(&As[cur][h][0], wm * 32 + 16, lane);
      const v16bf b  = frag_b_sw(&Bs[cur][h][wn * 512], lane);
      acc[0] = __builtin_amdgcn_wmma_f32_16x16x32_bf16(false, a0, false, b, (short)0, acc[0], false, false);
      acc[1] = __builtin_amdgcn_wmma_f32_16x16x32_bf16(false, a1, false, b, (short)0, acc[1], false, false);
    }
    if (kb2 + 1 < KB2) async_wait0();
    __syncthreads();
  }

  const int col = n0 + wn * 16 + (lane & 15);
  const float bv = bias[col];
#pragma unroll
  for (int sub = 0; sub < 2; ++sub) {
#pragma unroll
    for (int i = 0; i < 8; ++i) {
      const int row = row0 + wm * 32 + sub * 16 + ((lane >> 4) << 3) + i;
      float v = acc[sub][i] + bv;
      if (relu_to_bf16) {
        v = v > 0.0f ? v : 0.0f;
        outb[(size_t)row * ldob + col] = (__bf16)v;
      } else {
        outf[(size_t)row * ldof + col] = v;
      }
    }
  }
}

// ---------------------------------------------------------------------------
// Host-side orchestration
// ---------------------------------------------------------------------------
extern "C" void kernel_launch(void* const* d_in, const int* in_sizes, int n_in,
                              void* d_out, int out_size, void* d_ws, size_t ws_size,
                              hipStream_t stream) {
  (void)in_sizes; (void)n_in; (void)out_size; (void)ws_size;

  const float* prev_y      = (const float*)d_in[0];
  const float* prev_hidden = (const float*)d_in[1];
  const float* cur_c       = (const float*)d_in[2];
  const float* R_W         = (const float*)d_in[3];
  const float* IcW         = (const float*)d_in[4];
  const float* IfW         = (const float*)d_in[5];
  const float* O1W         = (const float*)d_in[6];
  const float* O1b         = (const float*)d_in[7];
  const float* O2W         = (const float*)d_in[8];
  const float* O2b         = (const float*)d_in[9];
  const float* O3W         = (const float*)d_in[10];
  const float* O3b         = (const float*)d_in[11];
  const float* O4W         = (const float*)d_in[12];
  const float* O4b         = (const float*)d_in[13];
  const float* bu          = (const float*)d_in[14];
  const float* br          = (const float*)d_in[15];
  const float* be          = (const float*)d_in[16];

  float* out        = (float*)d_out;
  float* out_coarse = out;
  float* out_fine   = out + (size_t)B_DIM * Q_DIM;
  float* hidden_f32 = out + (size_t)2 * B_DIM * Q_DIM;

  char* ws = (char*)d_ws;
  size_t off = 0;
  auto carve = [&](size_t bytes) {
    char* p = ws + off;
    off += (bytes + 255) & ~(size_t)255;
    return p;
  };
  __bf16* A16   = (__bf16*)carve((size_t)B_DIM * H_DIM * 2);      // prev_hidden bf16
  __bf16* RW16  = (__bf16*)carve((size_t)H_DIM * 3 * H_DIM * 2);  // R_W swizzled
  __bf16* O1W16 = (__bf16*)carve((size_t)S_DIM * S_DIM * 2);
  __bf16* O2W16 = (__bf16*)carve((size_t)S_DIM * Q_DIM * 2);
  __bf16* O3W16 = (__bf16*)carve((size_t)S_DIM * S_DIM * 2);
  __bf16* O4W16 = (__bf16*)carve((size_t)S_DIM * Q_DIM * 2);
  __bf16* H16   = (__bf16*)carve((size_t)B_DIM * H_DIM * 2);      // hidden bf16
  __bf16* Tc16  = (__bf16*)carve((size_t)B_DIM * S_DIM * 2);
  __bf16* Tf16  = (__bf16*)carve((size_t)B_DIM * S_DIM * 2);

  const dim3 blk(256);

  // 1) prev_hidden f32 -> bf16 (row-major, A operand)
  {
    const int n4 = (B_DIM * H_DIM) / 4;
    cvt_f32_bf16<<<dim3((n4 + 255) / 256), blk, 0, stream>>>(prev_hidden, A16, n4);
  }
  // 2) weights -> fragment-native bf16 layout
  auto swz = [&](const float* src, __bf16* dst, int K, int N) {
    const int nt = (N / 16) * (K / 32) * 32;
    swizzle_w_bf16<<<dim3((nt + 255) / 256), blk, 0, stream>>>(src, dst, K, N);
  };
  swz(R_W, RW16, H_DIM, 3 * H_DIM);
  swz(O1W, O1W16, S_DIM, S_DIM);
  swz(O2W, O2W16, S_DIM, Q_DIM);
  swz(O3W, O3W16, S_DIM, S_DIM);
  swz(O4W, O4W16, S_DIM, Q_DIM);

  // 3) fused GRU GEMM
  gru_gemm<<<dim3(B_DIM / 64, H_DIM / 64), blk, 0, stream>>>(
      A16, RW16, prev_y, cur_c, IcW, IfW, prev_hidden,
      bu, br, be, hidden_f32, H16);

  // 4) MLP heads layer 1 (ReLU -> bf16 scratch)
  mlp_gemm<<<dim3(B_DIM / 64, S_DIM / 64), blk, 0, stream>>>(
      H16,         H_DIM, O1W16, S_DIM / 64, O1b, 1, Tc16, S_DIM, nullptr, 0);
  mlp_gemm<<<dim3(B_DIM / 64, S_DIM / 64), blk, 0, stream>>>(
      H16 + S_DIM, H_DIM, O3W16, S_DIM / 64, O3b, 1, Tf16, S_DIM, nullptr, 0);

  // 5) MLP heads layer 2 (bias -> f32 output)
  mlp_gemm<<<dim3(B_DIM / 64, Q_DIM / 64), blk, 0, stream>>>(
      Tc16, S_DIM, O2W16, S_DIM / 64, O2b, 0, nullptr, 0, out_coarse, Q_DIM);
  mlp_gemm<<<dim3(B_DIM / 64, Q_DIM / 64), blk, 0, stream>>>(
      Tf16, S_DIM, O4W16, S_DIM / 64, O4b, 0, nullptr, 0, out_fine, Q_DIM);
}
